// Attention_4904852652841
// MI455X (gfx1250) — matrix operand order repaired and checked by reading, not compiled, not run on hardware
//
#include <hip/hip_runtime.h>
#include <math.h>

// Problem constants (reference: T=2048, E=1024, H=16, D=64)
constexpr int T  = 2048;
constexpr int E  = 1024;
constexpr int H  = 16;
constexpr int D  = 64;
constexpr int HD = H * D;   // 1024

typedef _Float16 f16;
typedef __attribute__((ext_vector_type(16))) _Float16 v16h;
typedef __attribute__((ext_vector_type(8)))  _Float16 v8h;
typedef __attribute__((ext_vector_type(8)))  float    v8f;
typedef __attribute__((ext_vector_type(4)))  unsigned u32x4;
typedef __attribute__((ext_vector_type(8)))  int      i32x8;
typedef __attribute__((ext_vector_type(4)))  int      i32x4;

union AV16 { v16h v; v8h h[2]; };

__device__ __forceinline__ v8f wmma_f16(v16h a, v16h b, v8f c) {
  // D = A(16x32 f16) * B(32x16 f16) + C(16x16 f32)
  return __builtin_amdgcn_wmma_f32_16x16x32_f16(
      /*neg_a=*/false, a, /*neg_b=*/false, b,
      /*c_mod=*/(short)0, c, /*reuse_a=*/false, /*reuse_b=*/false);
}

// ---------------------------------------------------------------------------
// Tensor Data Mover: 2-D tile (rows x tile_k f16) from global -> LDS,
// LDS written contiguously row-major [rows][tile_k].
// Descriptor packing per CDNA5 ISA.
// This toolchain exposes the 6-arg builtin:
//   (u32x4 group0, i32x8 group1, i32x4 group2, i32x4 group3, i32x8, i32 cpol)
// ---------------------------------------------------------------------------
__device__ __forceinline__ void tdm_load_2d(unsigned lds_off, const void* gptr,
                                            unsigned tile_k, unsigned tile_rows,
                                            unsigned row_stride_elems,
                                            unsigned tensor_d0,
                                            unsigned tensor_d1) {
  const unsigned long long ga = (unsigned long long)(uintptr_t)gptr;
  u32x4 g0;
  g0[0] = 1u;                                   // count=1, user mode, no gather
  g0[1] = lds_off;                              // lds_addr [63:32]
  g0[2] = (unsigned)(ga & 0xffffffffu);         // global_addr [95:64]
  g0[3] = (unsigned)((ga >> 32) & 0x1ffffffu)   // global_addr [120:96]
        | (2u << 30);                           // type=2 ("image") [127:126]
  i32x8 g1;
  g1[0] = (int)(1u << 16);                      // data_size=1 -> 2 bytes/elem
  g1[1] = (int)((tensor_d0 & 0xffffu) << 16);   // tensor_dim0[15:0] @ [63:48]
  g1[2] = (int)(((tensor_d0 >> 16) & 0xffffu) | ((tensor_d1 & 0xffffu) << 16));
  g1[3] = (int)(((tensor_d1 >> 16) & 0xffffu) | ((tile_k & 0xffffu) << 16));
  g1[4] = (int)(tile_rows & 0xffffu);           // tile_dim1; tile_dim2=0 (2-D)
  g1[5] = (int)row_stride_elems;                // tensor_dim0_stride[31:0]
  g1[6] = 0;                                    // stride[47:32]=0, dim1_stride=0
  g1[7] = 0;
  const i32x4 z4 = {0, 0, 0, 0};                // groups 2/3: unused for 2-D
  const i32x8 z8 = {0, 0, 0, 0, 0, 0, 0, 0};
  __builtin_amdgcn_tensor_load_to_lds(g0, g1, z4, z4, z8, 0);
}

// ---------------------------------------------------------------------------
// fp32 -> f16 convert (weights [H,D,E] flat are already Bt[N=h*D+d][K=e])
// ---------------------------------------------------------------------------
__global__ void cvt_f32_f16(const float* __restrict__ src,
                            f16* __restrict__ dst, int n) {
  int i = blockIdx.x * blockDim.x + threadIdx.x;
  if (i < n) dst[i] = (f16)src[i];
}

// V [T, HD] f16  ->  Vt [HD, T] f16  (so P*V B-fragments are contiguous)
__global__ void transpose_V(const f16* __restrict__ V, f16* __restrict__ Vt) {
  int i = blockIdx.x * blockDim.x + threadIdx.x;   // over HD*T
  if (i >= HD * T) return;
  int t  = i & (T - 1);
  int hd = i >> 11;       // i / T
  Vt[i] = V[(size_t)t * HD + hd];
}

// ---------------------------------------------------------------------------
// TDM-staged, double-buffered WMMA GEMM: C[M,N] = A[M,K] * Bt[N,K]^T (+bias).
// Block = 256 threads = 8 waves; block tile 64x128; per-wave 32x32 (4 WMMA
// per 32-deep k step, A frags reused over 2 n-tiles, B frags over 2 m-tiles).
// K chunk = 64; wave 0 issues two tensor_load_to_lds per chunk (A panel 8KB,
// B panel 16KB), overlapped with compute on the other buffer.
// grid = (N/128, M/64), dynamic LDS = 48KB.
// ---------------------------------------------------------------------------
constexpr int GM = 64, GN = 128, GK = 64;
constexpr int GEMM_BUF_BYTES = (GM + GN) * GK * 2;            // 24576
constexpr int GEMM_SMEM      = 2 * GEMM_BUF_BYTES;            // 49152

template <bool OUT_F32>
__global__ __launch_bounds__(256)
void gemm16(const f16* __restrict__ A, const f16* __restrict__ Bt,
            void* __restrict__ Cout, const float* __restrict__ bias,
            int M, int N, int Kdim) {
  extern __shared__ char gsm[];
  const int tid   = threadIdx.x;
  const int lane  = tid & 31;
  const int wave  = tid >> 5;
  const int lo    = lane & 15;
  const int hi    = lane >> 4;
  const int waveM = wave >> 2;        // 0..1
  const int waveN = wave & 3;         // 0..3
  const int m_blk = blockIdx.y * GM;
  const int n_blk = blockIdx.x * GN;
  const int mw    = waveM * 32;       // wave's rows within block tile
  const int nw    = waveN * 32;       // wave's cols within block tile

  const unsigned ldsBase = __builtin_amdgcn_groupstaticsize();

  auto issue = [&](int kc, int b) {
    const unsigned aoff = ldsBase + (unsigned)b * GEMM_BUF_BYTES;
    const unsigned boff = aoff + GM * GK * 2;
    tdm_load_2d(aoff, A + (size_t)m_blk * Kdim + kc, GK, GM,
                (unsigned)Kdim, (unsigned)Kdim, (unsigned)M);
    tdm_load_2d(boff, Bt + (size_t)n_blk * Kdim + kc, GK, GN,
                (unsigned)Kdim, (unsigned)Kdim, (unsigned)N);
  };

  v8f acc00 = {0.f, 0.f, 0.f, 0.f, 0.f, 0.f, 0.f, 0.f};
  v8f acc01 = acc00, acc10 = acc00, acc11 = acc00;

  if (wave == 0) {
    issue(0, 0);
    __builtin_amdgcn_s_wait_tensorcnt(0);
  }
  __syncthreads();

  for (int kc = 0; kc < Kdim; kc += GK) {
    const int cb = (kc >> 6) & 1;
    if (wave == 0 && kc + GK < Kdim) issue(kc + GK, cb ^ 1);

    const f16* Alds = (const f16*)(gsm + cb * GEMM_BUF_BYTES);
    const f16* Blds = Alds + GM * GK;

#pragma unroll
    for (int kk = 0; kk < GK; kk += 32) {
      AV16 a0, a1;
      const f16* ar0 = Alds + (mw + lo) * GK + kk;
      a0.h[0] = *(const v8h*)(ar0 + hi * 16);
      a0.h[1] = *(const v8h*)(ar0 + 8 + hi * 16);
      const f16* ar1 = Alds + (mw + 16 + lo) * GK + kk;
      a1.h[0] = *(const v8h*)(ar1 + hi * 16);
      a1.h[1] = *(const v8h*)(ar1 + 8 + hi * 16);

      v16h b0 = *(const v16h*)(Blds + (nw + lo) * GK + kk + hi * 16);
      v16h b1 = *(const v16h*)(Blds + (nw + 16 + lo) * GK + kk + hi * 16);

      acc00 = wmma_f16(a0.v, b0, acc00);
      acc01 = wmma_f16(a0.v, b1, acc01);
      acc10 = wmma_f16(a1.v, b0, acc10);
      acc11 = wmma_f16(a1.v, b1, acc11);
    }

    if (wave == 0) __builtin_amdgcn_s_wait_tensorcnt(0);
    __syncthreads();
  }

  // Epilogue: C/D layout row = j + 8*hi, col = lane&15 within each 16x16 tile
  auto store_tile = [&](v8f acc, int mt, int nt) {
#pragma unroll
    for (int j = 0; j < 8; ++j) {
      const int r = m_blk + mw + mt * 16 + j + hi * 8;
      const int c = n_blk + nw + nt * 16 + lo;
      if constexpr (OUT_F32)
        ((float*)Cout)[(size_t)r * N + c] = acc[j] + bias[c];
      else
        ((f16*)Cout)[(size_t)r * N + c] = (f16)acc[j];
    }
  };
  store_tile(acc00, 0, 0);
  store_tile(acc01, 0, 1);
  store_tile(acc10, 1, 0);
  store_tile(acc11, 1, 1);
}

// ---------------------------------------------------------------------------
// Attention core: one block (128 thr = 4 waves) per (16 q-rows, head).
// LDS: scores 16x2048 f32 (128KB) + p16 16x2048 f16 (64KB) + reductions.
// Phase 1: S = Q K^T / 8 with causal mask (WMMA, full row band in LDS)
// Phase 2: row softmax in LDS; stream probs (fp32) to HBM once; keep f16 copy
// Phase 3: O = P V via WMMA (A-fragments from LDS, B from Vt) -> concat f16
// ---------------------------------------------------------------------------
constexpr int SCORE_BYTES = 16 * T * 4;
constexpr int P16_BYTES   = 16 * T * 2;
constexpr int ATTN_SMEM   = SCORE_BYTES + P16_BYTES + (128 + 32) * 4;

__global__ __launch_bounds__(128)
void attn_core(const f16* __restrict__ Qf, const f16* __restrict__ Kf,
               const f16* __restrict__ Vt, f16* __restrict__ concat,
               float* __restrict__ probs) {
  extern __shared__ char smem[];
  float* scores = (float*)smem;                                // [16][T]
  f16*   p16    = (f16*)(smem + SCORE_BYTES);                  // [16][T]
  float* red    = (float*)(smem + SCORE_BYTES + P16_BYTES);    // [16*8]
  float* rowmax = red + 128;                                   // [16]
  float* rowinv = rowmax + 16;                                 // [16]

  const int tid  = threadIdx.x;
  const int lane = tid & 31;
  const int wave = tid >> 5;
  const int lo   = lane & 15;
  const int hi   = lane >> 4;

  const int qrow0 = blockIdx.x * 16;
  const int h     = blockIdx.y;

  // ---- Phase 1: scores -------------------------------------------------
  const f16* qbase = Qf + (size_t)(qrow0 + lo) * HD + h * D;
  AV16 aq0, aq1;
  aq0.h[0] = *(const v8h*)(qbase + hi * 16);
  aq0.h[1] = *(const v8h*)(qbase + 8 + hi * 16);
  aq1.h[0] = *(const v8h*)(qbase + 32 + hi * 16);
  aq1.h[1] = *(const v8h*)(qbase + 40 + hi * 16);

  for (int st = wave; st < T / 16; st += 4) {
    const int s0 = st * 16;
    const f16* kb = Kf + (size_t)(s0 + lo) * HD + h * D;
    v16h b0 = *(const v16h*)(kb + hi * 16);        // K(d) = 0..31
    v16h b1 = *(const v16h*)(kb + 32 + hi * 16);   // K(d) = 32..63
    v8f acc = {0.f, 0.f, 0.f, 0.f, 0.f, 0.f, 0.f, 0.f};
    acc = wmma_f16(aq0.v, b0, acc);
    acc = wmma_f16(aq1.v, b1, acc);

    const int col = s0 + lo;
    for (int j = 0; j < 8; ++j) {
      const int r = j + hi * 8;                    // row within band
      float v = acc[j] * 0.125f;                   // 1/sqrt(64)
      if (col > qrow0 + r) v = -INFINITY;          // causal mask
      scores[r * T + col] = v;
    }
  }
  __syncthreads();

  // ---- Phase 2: softmax over each 2048-wide row ------------------------
  const int row = tid & 15;   // 0..15
  const int sub = tid >> 4;   // 0..7
  float m = -INFINITY;
  for (int c = sub; c < T; c += 8) m = fmaxf(m, scores[row * T + c]);
  red[row * 8 + sub] = m;
  __syncthreads();
  if (sub == 0) {
    float mm = red[row * 8];
    for (int j = 1; j < 8; ++j) mm = fmaxf(mm, red[row * 8 + j]);
    rowmax[row] = mm;
  }
  __syncthreads();
  const float rm = rowmax[row];
  float s = 0.f;
  for (int c = sub; c < T; c += 8) {
    float e = __expf(scores[row * T + c] - rm);
    scores[row * T + c] = e;
    s += e;
  }
  red[row * 8 + sub] = s;
  __syncthreads();
  if (sub == 0) {
    float ss = 0.f;
    for (int j = 0; j < 8; ++j) ss += red[row * 8 + j];
    rowinv[row] = 1.0f / ss;
  }
  __syncthreads();

  // stream probs (coalesced, written exactly once) + keep f16 copy in LDS
  for (int i = tid; i < 16 * T; i += 128) {
    const int r = i >> 11;         // / T
    const int c = i & (T - 1);
    const float p = scores[r * T + c] * rowinv[r];
    probs[((size_t)h * T + (qrow0 + r)) * T + c] = p;
    p16[r * T + c] = (f16)p;
  }
  __syncthreads();

  // ---- Phase 3: O = P * V (K = T = 2048), one 16x16 n-tile per wave ----
  const int n0 = wave * 16;   // D = 64 -> 4 tiles, 4 waves
  const f16* vtb = Vt + ((size_t)h * D + n0 + lo) * T;
  v8f acc = {0.f, 0.f, 0.f, 0.f, 0.f, 0.f, 0.f, 0.f};
  for (int kc = 0; kc < T; kc += 32) {
    AV16 a;
    a.h[0] = *(const v8h*)(&p16[lo * T + kc + hi * 16]);
    a.h[1] = *(const v8h*)(&p16[lo * T + kc + 8 + hi * 16]);
    v16h b = *(const v16h*)(vtb + kc + hi * 16);
    acc = wmma_f16(a.v, b, acc);
  }
  for (int j = 0; j < 8; ++j) {
    const int r = qrow0 + j + hi * 8;
    concat[(size_t)r * HD + h * D + n0 + lo] = (f16)acc[j];
  }
}

// ---------------------------------------------------------------------------
extern "C" void kernel_launch(void* const* d_in, const int* in_sizes, int n_in,
                              void* d_out, int out_size, void* d_ws,
                              size_t ws_size, hipStream_t stream) {
  const float* seq = (const float*)d_in[0];   // [T,E]
  const float* Wq  = (const float*)d_in[1];   // [H,D,E] == Bt[HD][E]
  const float* Wk  = (const float*)d_in[2];
  const float* Wv  = (const float*)d_in[3];
  const float* Wo  = (const float*)d_in[4];   // [E,HD]  == Bt[E][HD]
  const float* bo  = (const float*)d_in[5];   // [E]

  float* out   = (float*)d_out;               // [T,E]
  float* probs = out + (size_t)T * E;         // [H,T,T]

  // Workspace layout (f16), ~32 MB total
  char* ws = (char*)d_ws;
  size_t off = 0;
  auto alloc = [&](size_t elems) {
    f16* p = (f16*)(ws + off);
    off += ((elems * sizeof(f16) + 255) & ~(size_t)255);
    return p;
  };
  f16* seqh   = alloc((size_t)T * E);
  f16* Wqh    = alloc((size_t)HD * E);
  f16* Wkh    = alloc((size_t)HD * E);
  f16* Wvh    = alloc((size_t)HD * E);
  f16* Woh    = alloc((size_t)E * HD);
  f16* Qh     = alloc((size_t)T * HD);
  f16* Kh     = alloc((size_t)T * HD);
  f16* Vh     = alloc((size_t)T * HD);
  f16* Vth    = alloc((size_t)HD * T);
  f16* concat = alloc((size_t)T * HD);

  // 1) fp32 -> f16 converts
  auto cvt = [&](const float* s, f16* d, int n) {
    cvt_f32_f16<<<(n + 255) / 256, 256, 0, stream>>>(s, d, n);
  };
  cvt(seq, seqh, T * E);
  cvt(Wq, Wqh, HD * E);
  cvt(Wk, Wkh, HD * E);
  cvt(Wv, Wvh, HD * E);
  cvt(Wo, Woh, E * HD);

  // 2) Q/K/V projections: [T,E] x [HD,E]^T -> [T,HD] f16 (TDM-staged GEMM)
  dim3 gProj(HD / GN, T / GM);
  gemm16<false><<<gProj, 256, GEMM_SMEM, stream>>>(seqh, Wqh, Qh, nullptr,
                                                   T, HD, E);
  gemm16<false><<<gProj, 256, GEMM_SMEM, stream>>>(seqh, Wkh, Kh, nullptr,
                                                   T, HD, E);
  gemm16<false><<<gProj, 256, GEMM_SMEM, stream>>>(seqh, Wvh, Vh, nullptr,
                                                   T, HD, E);

  // 3) V transpose for contiguous P*V B-fragments
  transpose_V<<<(HD * T + 255) / 256, 256, 0, stream>>>(Vh, Vth);

  // 4) attention core (scores + softmax + probs + P*V)
  static bool attr_set = [] {
    return hipFuncSetAttribute((const void*)attn_core,
                               hipFuncAttributeMaxDynamicSharedMemorySize,
                               ATTN_SMEM) == hipSuccess;
  }();
  (void)attr_set;
  attn_core<<<dim3(T / 16, H), 128, ATTN_SMEM, stream>>>(Qh, Kh, Vth, concat,
                                                         probs);

  // 5) output projection: [T,HD] x [E,HD]^T + bo -> [T,E] fp32
  dim3 gOut(E / GN, T / GM);
  gemm16<true><<<gOut, 256, GEMM_SMEM, stream>>>(concat, Woh, (void*)out, bo,
                                                 T, E, HD);
}
